// Corr_30485677868011
// MI455X (gfx1250) — compile-verified, matrix operand order checked by reading, not run on hardware
//
#include <hip/hip_runtime.h>

// Problem constants (match reference: x,y [4, 64, 256, 512] fp32, MAX_DISP=40)
#define BDIM   4
#define CCH    64
#define HDIM   256
#define WDIM   512
#define WSTRIP 128          // w-strip per workgroup
#define HALO   48           // 40 rounded up to 3 tiles of 16
#define XSPAN  176          // x strip: [WB, WB+176)
#define YSPAN  176          // y strip: [WB-48, WB+128)
#define NTHREADS 256
#define SCALE (1.0f / 5184.0f)   // 1 / (81 disparities * 64 channels)

typedef __attribute__((ext_vector_type(16))) _Float16 v16h;
typedef __attribute__((ext_vector_type(8)))  float    v8f;

#define WMMA(ACC, A, B) \
  ACC = __builtin_amdgcn_wmma_f32_16x16x32_f16(false, (A), false, (B), (short)0, (ACC), false, false)

__global__ __launch_bounds__(NTHREADS)
void corr_wmma_kernel(const float* __restrict__ x,
                      const float* __restrict__ y,
                      float* __restrict__ out)
{
  // f16 staging in [w][c] layout so each lane's WMMA operand (16 consecutive
  // channels at fixed w) is one aligned 32B LDS read.
  __shared__ __align__(32) _Float16 lx[XSPAN * CCH];
  __shared__ __align__(32) _Float16 ly[YSPAN * CCH];

  const int tid = threadIdx.x;
  const int b   = blockIdx.z;
  const int h   = blockIdx.y;
  const int WB  = blockIdx.x * WSTRIP;

  // ---------------- stage fp32 -> f16 into LDS (zero-fill out of range) ----
  for (int i = tid; i < XSPAN * CCH; i += NTHREADS) {
    const int wl = i % XSPAN;          // local w (lanes -> consecutive w: coalesced)
    const int c  = i / XSPAN;
    const size_t base = (((size_t)b * CCH + c) * HDIM + h) * (size_t)WDIM;

    const int gx = WB + wl;            // x strip [WB, WB+176)
    const float vx = (gx < WDIM) ? x[base + gx] : 0.0f;
    lx[wl * CCH + c] = (_Float16)vx;

    const int gy = WB - HALO + wl;     // y strip [WB-48, WB+128)
    const float vy = (gy >= 0) ? y[base + gy] : 0.0f;
    ly[wl * CCH + c] = (_Float16)vy;
  }
  __syncthreads();

  // ---------------- per-wave banded GEMM ----------------
  const int lane = tid & 31;
  const int wv   = tid >> 5;               // wave -> output tile I = WB/16 + wv
  const int wsel = lane & 15;              // operand row/col within tile
  const int ksel = (lane >> 4) << 4;       // K half select (A/B layouts)

  v8f accFull = {};   // x(I+1)*y(I)  [d=1..31]  +  y(I-1)*x(I)  [d=1..31], unmasked
  v8f accA    = {};   // x(I  )*y(I)  keep m>=n          (d = m-n   in [0,15])
  v8f accB    = {};   // y(I  )*x(I)  keep n> m          (d = n-m   in [1,15])
  v8f accC2   = {};   // x(I+2)*y(I)  keep m-n<=8        (d = 32+m-n <= 40)
  v8f accC3   = {};   // x(I+3)*y(I)  keep n-m>=8        (d = 48+m-n <= 40)
  v8f accE2   = {};   // y(I-2)*x(I)  keep n-m<=8        (d = 32+n-m <= 40)
  v8f accE3   = {};   // y(I-3)*x(I)  keep m-n>=8        (d = 48+n-m <= 40)

#pragma unroll
  for (int kh = 0; kh < 2; ++kh) {         // C=64 -> two K=32 WMMA steps
    const int kofs = kh * 32 + ksel;

    // x tiles I..I+3 (local w base 16*(wv+t)); y tiles I-3..I (local 16*wv+16*u)
    const v16h ax0 = *(const v16h*)(lx + (16 * (wv + 0) + wsel) * CCH + kofs);
    const v16h ax1 = *(const v16h*)(lx + (16 * (wv + 1) + wsel) * CCH + kofs);
    const v16h ax2 = *(const v16h*)(lx + (16 * (wv + 2) + wsel) * CCH + kofs);
    const v16h ax3 = *(const v16h*)(lx + (16 * (wv + 3) + wsel) * CCH + kofs);
    const v16h ay0 = *(const v16h*)(ly + (16 * wv +  0 + wsel) * CCH + kofs);  // tile I-3
    const v16h ay1 = *(const v16h*)(ly + (16 * wv + 16 + wsel) * CCH + kofs);  // tile I-2
    const v16h ay2 = *(const v16h*)(ly + (16 * wv + 32 + wsel) * CCH + kofs);  // tile I-1
    const v16h ay3 = *(const v16h*)(ly + (16 * wv + 48 + wsel) * CCH + kofs);  // tile I

    // Z = X^T Y tiles: column sums give  sum_{d=0..40} <x[w+d], y[w]>
    WMMA(accA,    ax0, ay3);
    WMMA(accFull, ax1, ay3);
    WMMA(accC2,   ax2, ay3);
    WMMA(accC3,   ax3, ay3);
    // Zt = Y^T X tiles: column sums give  sum_{d=1..40} <x[w], y[w-d]>
    WMMA(accB,    ay3, ax0);
    WMMA(accFull, ay2, ax0);
    WMMA(accE2,   ay1, ax0);
    WMMA(accE3,   ay0, ax0);
  }

  // ---------------- masked column reduction ----------------
  // C/D layout: lanes 0-15 -> N=lane, M=r; lanes 16-31 -> N=lane-16, M=r+8.
  const int n     = lane & 15;
  const int mbase = (lane >> 4) << 3;
  float colsum = 0.0f;
#pragma unroll
  for (int r = 0; r < 8; ++r) {
    const int m = mbase + r;
    float cs = accFull[r];
    if (m >= n)     cs += accA[r];
    if (n >  m)     cs += accB[r];
    if (m - n <= 8) cs += accC2[r];
    if (n - m >= 8) cs += accC3[r];
    if (n - m <= 8) cs += accE2[r];
    if (m - n >= 8) cs += accE3[r];
    colsum += cs;
  }

  // combine the two M-halves of each column (lane n <-> lane n+16)
  const float total = (colsum + __shfl_xor(colsum, 16)) * SCALE;
  if (lane < 16) {
    out[((size_t)b * HDIM + h) * (size_t)WDIM + WB + wv * 16 + n] = total;
  }
}

extern "C" void kernel_launch(void* const* d_in, const int* in_sizes, int n_in,
                              void* d_out, int out_size, void* d_ws, size_t ws_size,
                              hipStream_t stream) {
  const float* x = (const float*)d_in[0];
  const float* y = (const float*)d_in[1];
  float* out = (float*)d_out;

  dim3 grid(WDIM / WSTRIP, HDIM, BDIM);   // (4, 256, 4)
  dim3 block(NTHREADS);
  corr_wmma_kernel<<<grid, block, 0, stream>>>(x, y, out);
}
